// GCNLinkPredictor_30923764531232
// MI455X (gfx1250) — matrix operand order repaired and checked by reading, not compiled, not run on hardware
//
#include <hip/hip_runtime.h>
#include <hip/hip_bf16.h>

typedef __attribute__((ext_vector_type(16))) __bf16    v16bf;
typedef __attribute__((ext_vector_type(8)))  float     v8f;
typedef __attribute__((ext_vector_type(4)))  unsigned  v4u;

#define D_FEAT 128

static __device__ __forceinline__ unsigned bfbits(float f) {
    // round-to-nearest-even fp32 -> bf16 bit pattern
    unsigned u = __builtin_bit_cast(unsigned, f);
    return (u + 0x7FFFu + ((u >> 16) & 1u)) >> 16;
}

// ---------------- elementwise helpers ----------------

__global__ void zero_f32_kernel(float* __restrict__ p, long long n) {
    long long i = (long long)blockIdx.x * blockDim.x + threadIdx.x;
    if (i < n) p[i] = 0.0f;
}

__global__ void deg_kernel(const long long* __restrict__ dst, float* __restrict__ deg, long long E) {
    long long i = (long long)blockIdx.x * blockDim.x + threadIdx.x;
    if (i < E) atomicAdd(&deg[dst[i]], 1.0f);
}

__global__ void dinv_kernel(float* __restrict__ deg_to_dinv, int N) {
    long long i = (long long)blockIdx.x * blockDim.x + threadIdx.x;
    if (i < N) deg_to_dinv[i] = rsqrtf(deg_to_dinv[i] + 1.0f);  // self-loop adds 1
}

// convert 2 fp32 -> packed 2x bf16 per thread (b64 in, b32 out; pure streaming)
__global__ void f32_to_bf16_kernel(const float* __restrict__ in, unsigned* __restrict__ out,
                                   long long n2) {
    long long i = (long long)blockIdx.x * blockDim.x + threadIdx.x;
    if (i < n2) {
        float2 v = ((const float2*)in)[i];
        out[i] = bfbits(v.x) | (bfbits(v.y) << 16);
    }
}

// Pre-pack W[128,128] f32 into bf16 B-fragments: for tileN(8) x kk(4) x lane(32),
// the lane's 16 bf16 values (8 K-pairs down column n) are stored contiguously (32B).
__global__ void pack_w_kernel(const float* __restrict__ W, unsigned* __restrict__ Wp) {
    int t = blockIdx.x * blockDim.x + threadIdx.x;   // 0..1023
    if (t >= 1024) return;
    int lane  = t & 31;
    int kk    = (t >> 5) & 3;
    int tileN = t >> 7;
    int n     = tileN * 16 + (lane & 15);
    int kbase = (lane >> 4) * 8 + kk * 32;           // B layout: lane covers K = {kb..kb+7, 16+kb..16+kb+7}, kb = half*8
    unsigned* dst = Wp + (size_t)t * 8;              // 8 dwords = 16 bf16
#pragma unroll
    for (int j = 0; j < 8; ++j) {
        unsigned lo = bfbits(W[(size_t)(kbase + 2 * j + (j >= 4 ? 8 : 0))     * D_FEAT + n]);
        unsigned hi = bfbits(W[(size_t)(kbase + 2 * j + (j >= 4 ? 8 : 0) + 1) * D_FEAT + n]);
        dst[j] = lo | (hi << 16);
    }
}

// ---------------- bf16 WMMA GEMM: Y[N,128] = A[N,128](bf16) @ W(prepacked bf16) ----------------
// One wave per 16x16 output tile; inner loop = 4 x b128 loads + 1 WMMA per K-step.
__global__ void __launch_bounds__(256) gemm_bf16_wmma_kernel(const __bf16* __restrict__ A,
                                                             const __bf16* __restrict__ Wp,
                                                             float* __restrict__ Y, int N) {
    const int wave = threadIdx.x >> 5;
    const int lane = threadIdx.x & 31;
    const long long tile  = (long long)blockIdx.x * 8 + wave;
    const long long tileM = tile >> 3;
    const int       tileN = (int)(tile & 7);
    if (tileM * 16 >= N) return;

    const int m    = lane & 15;
    const int half = lane >> 4;
    // A fragment: lane covers K = {kb..kb+7, 16+kb..16+kb+7}, kb = half*8 (ISA 7.12.2)
    const __bf16* arow = A + (size_t)(tileM * 16 + m) * D_FEAT + half * 8;
    // B fragment: fully contiguous per lane thanks to pre-pack
    const __bf16* brow = Wp + ((size_t)tileN * 128 + lane) * 16;

    v8f acc = {};
#pragma unroll
    for (int kk = 0; kk < 4; ++kk) {
        union { v16bf bf; v4u u4[2]; } a, b;
        const __bf16* ap = arow + kk * 32;
        a.u4[0] = *(const v4u*)(ap);
        a.u4[1] = *(const v4u*)(ap + 16);
        const __bf16* bp = brow + (size_t)kk * 512;   // kk stride = 32 lanes * 16 elems
        b.u4[0] = *(const v4u*)(bp);
        b.u4[1] = *(const v4u*)(bp + 8);
        acc = __builtin_amdgcn_wmma_f32_16x16x32_bf16(
            /*neg_a=*/false, a.bf, /*neg_b=*/false, b.bf,
            /*c_mod=*/(short)0, acc, /*reuse_a=*/false, /*reuse_b=*/false);
    }

    // C/D layout: VGPR r -> M = r + half*8, N = lane&15
    float* yrow = Y + (size_t)(tileM * 16 + half * 8) * D_FEAT + tileN * 16 + m;
#pragma unroll
    for (int r = 0; r < 8; ++r)
        yrow[(size_t)r * D_FEAT] = acc[r];
}

// ---------------- edge scatter: agg[dst] += h[src] * dinv[src]*dinv[dst] ----------------
__global__ void __launch_bounds__(256) scatter_kernel(const float* __restrict__ h,
                                                      const long long* __restrict__ src,
                                                      const long long* __restrict__ dst,
                                                      const float* __restrict__ dinv,
                                                      float* __restrict__ agg, long long E) {
    long long wave = (long long)blockIdx.x * (blockDim.x >> 5) + (threadIdx.x >> 5);
    int lane = threadIdx.x & 31;
    if (wave >= E) return;
    long long s = src[wave], d = dst[wave];
    float norm = dinv[s] * dinv[d];
    const float* hs = h + (size_t)s * D_FEAT;
    float*       ad = agg + (size_t)d * D_FEAT;
#pragma unroll
    for (int j = 0; j < 4; ++j) {
        int f = lane + 32 * j;
        atomicAdd(&ad[f], hs[f] * norm);
    }
}

// layer-1 combine: relu(agg + h*dinv^2 + b1) written directly as bf16 (feeds GEMM2 only)
__global__ void combine_relu_bf16_kernel(const float* __restrict__ h, const float* __restrict__ agg,
                                         const float* __restrict__ b, const float* __restrict__ dinv,
                                         unsigned* __restrict__ outbf, int N) {
    long long i = (long long)blockIdx.x * blockDim.x + threadIdx.x;   // over NF/2
    if (i >= (long long)N * (D_FEAT / 2)) return;
    long long e0 = 2 * i;
    int node = (int)(e0 >> 7);
    int f = (int)(e0 & (D_FEAT - 1));
    float di = dinv[node];
    float d2 = di * di;
    float v0 = fmaxf(agg[e0]     + h[e0]     * d2 + b[f],     0.0f);
    float v1 = fmaxf(agg[e0 + 1] + h[e0 + 1] * d2 + b[f + 1], 0.0f);
    outbf[i] = bfbits(v0) | (bfbits(v1) << 16);
}

// layer-2 combine: z = agg + h*dinv^2 + b2, in place into agg (f32)
__global__ void combine_f32_kernel(const float* __restrict__ h, const float* __restrict__ b,
                                   const float* __restrict__ dinv, float* __restrict__ agg, int N) {
    long long idx = (long long)blockIdx.x * blockDim.x + threadIdx.x;
    if (idx >= (long long)N * D_FEAT) return;
    int node = (int)(idx >> 7);
    int f = (int)(idx & (D_FEAT - 1));
    float di = dinv[node];
    agg[idx] = agg[idx] + h[idx] * (di * di) + b[f];
}

// ---------------- pair head: out[p,:2] = [z_i ; z_j] @ lin_w^T + lin_b ----------------
__global__ void __launch_bounds__(256) pairs_kernel(const float* __restrict__ z,
                                                    const long long* __restrict__ pi,
                                                    const long long* __restrict__ pj,
                                                    const float* __restrict__ lin_w,
                                                    const float* __restrict__ lin_b,
                                                    float* __restrict__ out, long long P) {
    long long wave = (long long)blockIdx.x * (blockDim.x >> 5) + (threadIdx.x >> 5);
    int lane = threadIdx.x & 31;
    if (wave >= P) return;
    const float* zi = z + (size_t)pi[wave] * D_FEAT;
    const float* zj = z + (size_t)pj[wave] * D_FEAT;
    float a0 = 0.0f, a1 = 0.0f;
#pragma unroll
    for (int j = 0; j < 4; ++j) {
        int f = lane + 32 * j;
        float vi = zi[f], vj = zj[f];
        a0 += vi * lin_w[f]       + vj * lin_w[128 + f];
        a1 += vi * lin_w[256 + f] + vj * lin_w[384 + f];
    }
#pragma unroll
    for (int off = 16; off >= 1; off >>= 1) {
        a0 += __shfl_xor(a0, off, 32);
        a1 += __shfl_xor(a1, off, 32);
    }
    if (lane == 0) {
        out[wave * 2 + 0] = a0 + lin_b[0];
        out[wave * 2 + 1] = a1 + lin_b[1];
    }
}

extern "C" void kernel_launch(void* const* d_in, const int* in_sizes, int n_in,
                              void* d_out, int out_size, void* d_ws, size_t ws_size,
                              hipStream_t stream) {
    const float*     x     = (const float*)d_in[0];
    const long long* ei    = (const long long*)d_in[1];   // [2,E] int64: row0=src, row1=dst
    const long long* ep    = (const long long*)d_in[2];   // [2,P] int64
    const float*     W1    = (const float*)d_in[3];
    const float*     b1    = (const float*)d_in[4];
    const float*     W2    = (const float*)d_in[5];
    const float*     b2    = (const float*)d_in[6];
    const float*     lin_w = (const float*)d_in[7];       // [2,256]
    const float*     lin_b = (const float*)d_in[8];       // [2]
    float*           out   = (float*)d_out;

    const int       N = in_sizes[0] / D_FEAT;             // 100000
    const long long E = in_sizes[1] / 2;                  // 1.6M
    const long long P = in_sizes[2] / 2;                  // 200k
    const long long NF = (long long)N * D_FEAT;

    // Workspace: dinv | Wp1 | Wp2 | bf buffer (x-bf16, later relu-bf16) | bufA f32 | bufB f32
    char*  ws = (char*)d_ws;
    size_t o  = 0;
    float*  dinv = (float*)(ws + o);              o += ((size_t)N * 4 + 255) & ~(size_t)255;
    __bf16* Wp1  = (__bf16*)(ws + o);             o += 1024 * 16 * 2;     // 32KB
    __bf16* Wp2  = (__bf16*)(ws + o);             o += 1024 * 16 * 2;     // 32KB
    __bf16* bfb  = (__bf16*)(ws + o);             o += (size_t)NF * 2;    // 25.6MB
    float*  bufA = (float*)(ws + o);              o += (size_t)NF * 4;    // 51.2MB
    float*  bufB = (float*)(ws + o);

    const unsigned gNF   = (unsigned)((NF + 255) / 256);
    const unsigned gNF2  = (unsigned)((NF / 2 + 255) / 256);
    const unsigned gN    = (unsigned)((N + 255) / 256);
    const unsigned gE    = (unsigned)((E + 255) / 256);
    const unsigned gEw   = (unsigned)((E + 7) / 8);       // wave per edge, 8 waves/block
    const unsigned gPw   = (unsigned)((P + 7) / 8);
    const unsigned gGemm = (unsigned)((N + 15) / 16);     // 16-row block per 256-thread block

    // --- normalization: deg -> dinv = rsqrt(deg+1) ---
    zero_f32_kernel<<<gN, 256, 0, stream>>>(dinv, N);
    deg_kernel<<<gE, 256, 0, stream>>>(ei + E, dinv, E);
    dinv_kernel<<<gN, 256, 0, stream>>>(dinv, N);

    // --- one-time conversions/packs ---
    f32_to_bf16_kernel<<<gNF2, 256, 0, stream>>>(x, (unsigned*)bfb, NF / 2);
    pack_w_kernel<<<4, 256, 0, stream>>>(W1, (unsigned*)Wp1);
    pack_w_kernel<<<4, 256, 0, stream>>>(W2, (unsigned*)Wp2);

    // --- layer 1: h = x@W1 ; agg = scatter(h) ; bfb = bf16(relu(agg + h*dinv^2 + b1)) ---
    gemm_bf16_wmma_kernel<<<gGemm, 256, 0, stream>>>(bfb, Wp1, bufA, N);
    zero_f32_kernel<<<gNF, 256, 0, stream>>>(bufB, NF);
    scatter_kernel<<<gEw, 256, 0, stream>>>(bufA, ei, ei + E, dinv, bufB, E);
    combine_relu_bf16_kernel<<<gNF2, 256, 0, stream>>>(bufA, bufB, b1, dinv, (unsigned*)bfb, N);

    // --- layer 2: h2 = relu1@W2 ; agg2 = scatter(h2) ; z = agg2 + h2*dinv^2 + b2 ---
    gemm_bf16_wmma_kernel<<<gGemm, 256, 0, stream>>>(bfb, Wp2, bufA, N);
    zero_f32_kernel<<<gNF, 256, 0, stream>>>(bufB, NF);
    scatter_kernel<<<gEw, 256, 0, stream>>>(bufA, ei, ei + E, dinv, bufB, E);
    combine_f32_kernel<<<gNF, 256, 0, stream>>>(bufA, b2, dinv, bufB, N);

    // --- link head ---
    pairs_kernel<<<gPw, 256, 0, stream>>>(bufB, ep, ep + P, lin_w, lin_b, out, P);
}
